// LigerLinearFusedCrossEntropyLoss_58832462020867
// MI455X (gfx1250) — compile-verified, matrix operand order checked by reading, not run on hardware
//
#include <hip/hip_runtime.h>
#include <hip/hip_bf16.h>
#include <math.h>

// ---------------------------------------------------------------------------
// Fused linear + cross-entropy (Liger-style) for MI455X / gfx1250 (wave32).
//   logits = X[N,D] @ W[V,D]^T  -> online log-sum-exp + target gather -> mean
// GEMM runs as bf16 WMMA (v_wmma_f32_16x16x32_bf16) with f32 accumulation.
// W is pre-cast to bf16 in workspace (one bandwidth-bound pass), so the hot
// loop streams half the bytes and does zero converts: per 2 WMMAs it issues
// just 2 global_load_b128 + 4 ds_load_b128. A per-column-tile barrier pins
// the A-fragment LDS loads inside the loop (prevents LICM->scratch spills).
// ---------------------------------------------------------------------------

#define N_ROWS   8192
#define D_DIM    4096
#define V_DIM    32000
#define IGNORE_INDEX (-100)

#define M_TILE   32          // rows per workgroup (resident in LDS as bf16)
#define GRID_X   (N_ROWS / M_TILE)   // 256
#define GRID_Y   25                  // vocab splits
#define V_CHUNK  (V_DIM / GRID_Y)    // 1280
#define N_TILE   128                 // vocab cols per iteration (8 waves x 16)
#define ITERS    (V_CHUNK / N_TILE)  // 10

#define LDS_STRIDE 4104      // D_DIM + 8 bf16 pad -> 8208 B row stride (bank-friendly)
#define A_BYTES   (M_TILE * LDS_STRIDE * 2)          // 262,656 B
#define PART_F    (8 * M_TILE)                       // 8 waves x 32 rows
#define SMEM_BYTES (A_BYTES + 3 * PART_F * 4)        // + pm/ps/pt

typedef __attribute__((ext_vector_type(16))) __bf16 v16bf;
typedef __attribute__((ext_vector_type(8)))  __bf16 v8bf;
typedef __attribute__((ext_vector_type(4)))  __bf16 v4bf;
typedef __attribute__((ext_vector_type(8)))  float  v8f;

#define NEG_BIG (-3.0e38f)

// ---- one-pass fp32 -> bf16 cast of the weight matrix (bandwidth-bound) ----
__global__ __launch_bounds__(256)
void cast_w_kernel(const float* __restrict__ W, __bf16* __restrict__ Wb)
{
    const size_t total = (size_t)V_DIM * D_DIM;
    const size_t stride = (size_t)gridDim.x * blockDim.x * 4;
    for (size_t i = ((size_t)blockIdx.x * blockDim.x + threadIdx.x) * 4;
         i < total; i += stride) {
        const float4 f = *(const float4*)(W + i);
        v4bf o;
        o[0] = (__bf16)f.x; o[1] = (__bf16)f.y;
        o[2] = (__bf16)f.z; o[3] = (__bf16)f.w;
        *(v4bf*)(Wb + i) = o;
    }
}

// WBF16 = true: B comes from pre-cast bf16 weights; false: convert on the fly.
template <bool WBF16>
__global__ __launch_bounds__(256)
void flce_main_kernel(const float* __restrict__ X,
                      const float* __restrict__ W,      // fp32 weights
                      const __bf16* __restrict__ Wb,    // bf16 weights (or null)
                      const int* __restrict__ target,
                      float* __restrict__ ms_out,   // [N][GRID_Y][2] (max, sumexp)
                      float* __restrict__ tgt_out)  // [N] target logit
{
    extern __shared__ char smem[];
    __bf16* As = (__bf16*)smem;
    float*  pm = (float*)(smem + A_BYTES);
    float*  ps = pm + PART_F;
    float*  pt = ps + PART_F;

    const int tid   = threadIdx.x;
    const int row0  = blockIdx.x * M_TILE;
    const int v_lo  = blockIdx.y * V_CHUNK;

    // ---- stage A tile: 32 rows x 4096 fp32 -> bf16 in LDS (padded rows) ----
    for (int i = tid * 4; i < M_TILE * D_DIM; i += 256 * 4) {
        const int r = i >> 12;          // / 4096
        const int c = i & (D_DIM - 1);
        const float4 f = *(const float4*)(X + (size_t)(row0 + r) * D_DIM + c);
        __bf16* dst = As + r * LDS_STRIDE + c;
        dst[0] = (__bf16)f.x; dst[1] = (__bf16)f.y;
        dst[2] = (__bf16)f.z; dst[3] = (__bf16)f.w;
    }
    __syncthreads();

    const int lane = tid & 31;
    const int wave = tid >> 5;        // col group (0..7): cols 16*wave..+16
    const int half = lane >> 4;       // K-run selector per ISA 16-bit layout
    const int l16  = lane & 15;

    // A fragment bases: group g rows = 16*g + l16
    const __bf16* a_row0 = As + l16 * LDS_STRIDE;
    const __bf16* a_row1 = As + (16 + l16) * LDS_STRIDE;

    // element (g, j) of the two accumulators maps to row 16*g + j + 8*half
    int trow[2][8];
    #pragma unroll
    for (int g = 0; g < 2; ++g)
        #pragma unroll
        for (int j = 0; j < 8; ++j)
            trow[g][j] = target[row0 + 16 * g + j + 8 * half];

    float m_run[2][8], s_run[2][8], t_acc[2][8];
    #pragma unroll
    for (int g = 0; g < 2; ++g)
        #pragma unroll
        for (int j = 0; j < 8; ++j) {
            m_run[g][j] = NEG_BIG; s_run[g][j] = 0.f; t_acc[g][j] = 0.f;
        }

    for (int it = 0; it < ITERS; ++it) {
        const int vcol = v_lo + it * N_TILE + wave * 16 + l16;  // lane's vocab column

        const float*  wrow_f = W  + (size_t)vcol * D_DIM;
        const __bf16* wrow_b = Wb + (size_t)vcol * D_DIM;

        v8f acc0 = {};
        v8f acc1 = {};
        #pragma unroll 4
        for (int k = 0; k < D_DIM; k += 32) {
            const int off1 = k + 8 * half;

            // B fragment: two 8-element K-runs for this lane's vocab column
            v16bf b;
            if (WBF16) {
                v8bf blo = *(const v8bf*)(wrow_b + off1);        // 16B
                v8bf bhi = *(const v8bf*)(wrow_b + off1 + 16);   // 16B
                #pragma unroll
                for (int e = 0; e < 8; ++e) { b[e] = blo[e]; b[e + 8] = bhi[e]; }
            } else {
                const float4 f0 = *(const float4*)(wrow_f + off1);
                const float4 f1 = *(const float4*)(wrow_f + off1 + 4);
                const float4 f2 = *(const float4*)(wrow_f + off1 + 16);
                const float4 f3 = *(const float4*)(wrow_f + off1 + 20);
                b[0]=(__bf16)f0.x;  b[1]=(__bf16)f0.y;  b[2]=(__bf16)f0.z;  b[3]=(__bf16)f0.w;
                b[4]=(__bf16)f1.x;  b[5]=(__bf16)f1.y;  b[6]=(__bf16)f1.z;  b[7]=(__bf16)f1.w;
                b[8]=(__bf16)f2.x;  b[9]=(__bf16)f2.y;  b[10]=(__bf16)f2.z; b[11]=(__bf16)f2.w;
                b[12]=(__bf16)f3.x; b[13]=(__bf16)f3.y; b[14]=(__bf16)f3.z; b[15]=(__bf16)f3.w;
            }

            // A fragments from LDS: two 16B runs each (bank-conflict-free)
            v8bf a0lo = *(const v8bf*)(a_row0 + off1);
            v8bf a0hi = *(const v8bf*)(a_row0 + off1 + 16);
            v16bf a0;
            #pragma unroll
            for (int e = 0; e < 8; ++e) { a0[e] = a0lo[e]; a0[e + 8] = a0hi[e]; }

            acc0 = __builtin_amdgcn_wmma_f32_16x16x32_bf16(
                false, a0, false, b, (short)0, acc0, false, false);

            v8bf a1lo = *(const v8bf*)(a_row1 + off1);
            v8bf a1hi = *(const v8bf*)(a_row1 + off1 + 16);
            v16bf a1;
            #pragma unroll
            for (int e = 0; e < 8; ++e) { a1[e] = a1lo[e]; a1[e + 8] = a1hi[e]; }

            acc1 = __builtin_amdgcn_wmma_f32_16x16x32_bf16(
                false, a1, false, b, (short)0, acc1, false, false);
        }

        // online softmax update (per lane element; each covers one logit)
        #pragma unroll
        for (int g = 0; g < 2; ++g) {
            #pragma unroll
            for (int j = 0; j < 8; ++j) {
                const float x  = (g == 0) ? acc0[j] : acc1[j];
                const float nm = fmaxf(m_run[g][j], x);
                s_run[g][j] = s_run[g][j] * __expf(m_run[g][j] - nm) + __expf(x - nm);
                m_run[g][j] = nm;
                t_acc[g][j] += (trow[g][j] == vcol) ? x : 0.f;
            }
        }

        // LDS fence: keeps the A-fragment ds_loads inside this loop (no LICM
        // across column tiles -> no scratch spills) and keeps waves converged.
        __syncthreads();
    }

    // ---- merge across the 16 column lanes (halves stay independent) ----
    #pragma unroll
    for (int mask = 1; mask <= 8; mask <<= 1) {
        #pragma unroll
        for (int g = 0; g < 2; ++g) {
            #pragma unroll
            for (int j = 0; j < 8; ++j) {
                const float mo = __shfl_xor(m_run[g][j], mask, 32);
                const float so = __shfl_xor(s_run[g][j], mask, 32);
                const float to = __shfl_xor(t_acc[g][j], mask, 32);
                const float nm = fmaxf(m_run[g][j], mo);
                s_run[g][j] = s_run[g][j] * __expf(m_run[g][j] - nm)
                            + so * __expf(mo - nm);
                m_run[g][j] = nm;
                t_acc[g][j] += to;
            }
        }
    }

    // lanes 0 and 16 hold reduced rows; publish per-wave partials to LDS
    if (l16 == 0) {
        #pragma unroll
        for (int g = 0; g < 2; ++g) {
            #pragma unroll
            for (int j = 0; j < 8; ++j) {
                const int rwg = 16 * g + j + 8 * half;
                pm[wave * M_TILE + rwg] = m_run[g][j];
                ps[wave * M_TILE + rwg] = s_run[g][j];
                pt[wave * M_TILE + rwg] = t_acc[g][j];
            }
        }
    }
    __syncthreads();

    // ---- combine the 8 column-slice waves; write per-(row, vsplit) partial ----
    if (tid < M_TILE) {
        float M = NEG_BIG, S = 0.f, T = 0.f;
        #pragma unroll
        for (int c = 0; c < 8; ++c) {
            const float m = pm[c * M_TILE + tid];
            const float s = ps[c * M_TILE + tid];
            const float nm = fmaxf(M, m);
            S = S * __expf(M - nm) + s * __expf(m - nm);
            M = nm;
            T += pt[c * M_TILE + tid];
        }
        const int rg = row0 + tid;
        ms_out[((size_t)rg * GRID_Y + blockIdx.y) * 2 + 0] = M;
        ms_out[((size_t)rg * GRID_Y + blockIdx.y) * 2 + 1] = S;
        const int tg = target[rg];
        if (tg >= v_lo && tg < v_lo + V_CHUNK)   // exactly one vsplit owns the target
            tgt_out[rg] = T;
    }
}

__global__ __launch_bounds__(1024)
void flce_reduce_kernel(const float* __restrict__ ms,
                        const float* __restrict__ tgt_logit,
                        const int* __restrict__ target,
                        float* __restrict__ out)
{
    __shared__ float red_s[32];
    __shared__ float red_c[32];

    float loss = 0.f;
    float cnt  = 0.f;
    for (int r = threadIdx.x; r < N_ROWS; r += 1024) {
        if (target[r] == IGNORE_INDEX) continue;
        float M = NEG_BIG, S = 0.f;
        #pragma unroll 5
        for (int p = 0; p < GRID_Y; ++p) {
            const float m = ms[((size_t)r * GRID_Y + p) * 2 + 0];
            const float s = ms[((size_t)r * GRID_Y + p) * 2 + 1];
            const float nm = fmaxf(M, m);
            S = S * __expf(M - nm) + s * __expf(m - nm);
            M = nm;
        }
        loss += (M + __logf(S)) - tgt_logit[r];
        cnt  += 1.f;
    }
    #pragma unroll
    for (int mask = 16; mask >= 1; mask >>= 1) {
        loss += __shfl_xor(loss, mask, 32);
        cnt  += __shfl_xor(cnt,  mask, 32);
    }
    const int wave = threadIdx.x >> 5;
    if ((threadIdx.x & 31) == 0) { red_s[wave] = loss; red_c[wave] = cnt; }
    __syncthreads();
    if (threadIdx.x == 0) {
        float ts = 0.f, tc = 0.f;
        #pragma unroll
        for (int w = 0; w < 32; ++w) { ts += red_s[w]; tc += red_c[w]; }
        out[0] = ts / fmaxf(tc, 1.f);
    }
}

extern "C" void kernel_launch(void* const* d_in, const int* in_sizes, int n_in,
                              void* d_out, int out_size, void* d_ws, size_t ws_size,
                              hipStream_t stream) {
    (void)in_sizes; (void)n_in; (void)out_size;

    const float* X = (const float*)d_in[0];     // [N, D] fp32
    const float* W = (const float*)d_in[1];     // [V, D] fp32
    const int*   T = (const int*)d_in[2];       // [N]

    // workspace layout: [ms partials][tgt logits][bf16 weight copy]
    const size_t ms_f   = (size_t)N_ROWS * GRID_Y * 2;
    const size_t base_b = ms_f * 4 + (size_t)N_ROWS * 4;
    const size_t wb_b   = (size_t)V_DIM * D_DIM * 2;

    float*  ms  = (float*)d_ws;
    float*  tgt = ms + ms_f;
    __bf16* Wb  = (__bf16*)((char*)d_ws + base_b);

    const bool use_bf16_w = (ws_size >= base_b + wb_b);

    static bool attr_set = false;
    if (!attr_set) {
        hipFuncSetAttribute((const void*)flce_main_kernel<true>,
                            hipFuncAttributeMaxDynamicSharedMemorySize, SMEM_BYTES);
        hipFuncSetAttribute((const void*)flce_main_kernel<false>,
                            hipFuncAttributeMaxDynamicSharedMemorySize, SMEM_BYTES);
        attr_set = true;
    }

    dim3 grid(GRID_X, GRID_Y);
    if (use_bf16_w) {
        cast_w_kernel<<<4096, 256, 0, stream>>>(W, Wb);
        flce_main_kernel<true><<<grid, 256, SMEM_BYTES, stream>>>(X, W, Wb, T, ms, tgt);
    } else {
        flce_main_kernel<false><<<grid, 256, SMEM_BYTES, stream>>>(X, W, Wb, T, ms, tgt);
    }
    flce_reduce_kernel<<<1, 1024, 0, stream>>>(ms, tgt, T, (float*)d_out);
}
